// WindowAttentionBlock_77094662963556
// MI455X (gfx1250) — matrix operand order repaired and checked by reading, not compiled, not run on hardware
//
#include <hip/hip_runtime.h>
#include <hip/hip_bf16.h>
#include <stdint.h>

typedef __attribute__((ext_vector_type(16))) _Float16 v16h;
typedef __attribute__((ext_vector_type(8)))  float    v8f;
typedef __attribute__((ext_vector_type(4)))  uint32_t v4u;

union Frag {
    v16h     h;
    v4u      q[2];
    uint32_t u[8];
    _Float16 e[16];
};

#define TOK   262144      // 4 * 256 * 256 tokens
#define CDIM  192
#define HEADS 8
#define HD    24
#define NWIN  1024        // (256/8)^2 windows per batch image

// Optional CDNA5 async global->LDS path (compile-safe fallback if absent)
#if defined(__has_builtin)
#if __has_builtin(__builtin_amdgcn_global_load_async_to_lds_b128) && \
    __has_builtin(__builtin_amdgcn_s_wait_asynccnt)
#define HAVE_ASYNC_LDS 1
// builtin signature: (v4i addrspace(1)*, v4i addrspace(3)*, Imm offset, Imm cpol)
typedef int v4i_gcc __attribute__((vector_size(16)));
typedef __attribute__((address_space(1))) v4i_gcc* as1_v4i;
typedef __attribute__((address_space(3))) v4i_gcc* as3_v4i;
#endif
#endif

// ---------------------------------------------------------------------------
// Weight convert: w (K x N, f32) -> wt (N x K, f16)   wt[n*K+k] = w[k*N+n]
// ---------------------------------------------------------------------------
__global__ void wconv_kernel(const float* __restrict__ w, _Float16* __restrict__ wt,
                             int K, int N) {
    int idx = blockIdx.x * 256 + threadIdx.x;
    if (idx < K * N) {
        int n = idx / K, k = idx - n * K;
        wt[idx] = (_Float16)w[(size_t)k * N + n];
    }
}

// ---------------------------------------------------------------------------
// Precompute attention bias + shift-mask table:
//   biasTab[cls][head][nq][nk],  cls = (wh==31)*2 + (ww==31)
// ---------------------------------------------------------------------------
__global__ void bias_prep_kernel(const float* __restrict__ rpb, float* __restrict__ tab) {
    int idx = blockIdx.x * 256 + threadIdx.x;      // 131072 total
    int nk = idx & 63, nq = (idx >> 6) & 63, hh = (idx >> 12) & 7, cls = idx >> 15;
    int rq = nq >> 3, cq = nq & 7, rk = nk >> 3, ck = nk & 7;
    float b = rpb[(size_t)((rq - rk + 7) * 15 + (cq - ck + 7)) * HEADS + hh];
    int whl = cls >> 1, wwl = cls & 1;
    int regq = (whl ? ((rq < 4) ? 3 : 6) : 0) + (wwl ? ((cq < 4) ? 1 : 2) : 0);
    int regk = (whl ? ((rk < 4) ? 3 : 6) : 0) + (wwl ? ((ck < 4) ? 1 : 2) : 0);
    tab[idx] = b + ((regq != regk) ? -100.f : 0.f);
}

// ---------------------------------------------------------------------------
// LN1 + shift(-4,-4) + window partition.  One wave per token.
// ---------------------------------------------------------------------------
__global__ __launch_bounds__(256)
void ln1_kernel(const float* __restrict__ x, const float* __restrict__ g,
                const float* __restrict__ be, _Float16* __restrict__ h) {
    int wave = threadIdx.x >> 5, lane = threadIdx.x & 31;
    int t = blockIdx.x * 8 + wave;
    int n = t & 63, win = (t >> 6) & (NWIN - 1), b = t >> 16;
    int wh = win >> 5, ww = win & 31;
    int i = ((wh << 3) + (n >> 3) + 4) & 255;
    int j = ((ww << 3) + (n & 7) + 4) & 255;
    const float* xp = x + (size_t)b * CDIM * 65536 + (size_t)i * 256 + j;

    float vals[6];
    float sum = 0.f;
#pragma unroll
    for (int u = 0; u < 6; ++u) {
        vals[u] = xp[(size_t)(lane + 32 * u) * 65536];
        sum += vals[u];
    }
#pragma unroll
    for (int d = 16; d >= 1; d >>= 1) sum += __shfl_xor(sum, d, 32);
    float mean = sum * (1.f / 192.f);
    float var = 0.f;
#pragma unroll
    for (int u = 0; u < 6; ++u) { float df = vals[u] - mean; var += df * df; }
#pragma unroll
    for (int d = 16; d >= 1; d >>= 1) var += __shfl_xor(var, d, 32);
    float rs = rsqrtf(var * (1.f / 192.f) + 1e-5f);
#pragma unroll
    for (int u = 0; u < 6; ++u) {
        int ch = lane + 32 * u;
        h[(size_t)t * CDIM + ch] = (_Float16)((vals[u] - mean) * rs * g[ch] + be[ch]);
    }
}

// ---------------------------------------------------------------------------
// LN2: xh is (B,H,W,C) f32 contiguous per token.  One wave per token.
// ---------------------------------------------------------------------------
__global__ __launch_bounds__(256)
void ln2_kernel(const float* __restrict__ xh, const float* __restrict__ g,
                const float* __restrict__ be, _Float16* __restrict__ h) {
    int wave = threadIdx.x >> 5, lane = threadIdx.x & 31;
    int t = blockIdx.x * 8 + wave;
    const float* xp = xh + (size_t)t * CDIM;
    float vals[6];
    float sum = 0.f;
#pragma unroll
    for (int u = 0; u < 6; ++u) { vals[u] = xp[lane + 32 * u]; sum += vals[u]; }
#pragma unroll
    for (int d = 16; d >= 1; d >>= 1) sum += __shfl_xor(sum, d, 32);
    float mean = sum * (1.f / 192.f);
    float var = 0.f;
#pragma unroll
    for (int u = 0; u < 6; ++u) { float df = vals[u] - mean; var += df * df; }
#pragma unroll
    for (int d = 16; d >= 1; d >>= 1) var += __shfl_xor(var, d, 32);
    float rs = rsqrtf(var * (1.f / 192.f) + 1e-5f);
#pragma unroll
    for (int u = 0; u < 6; ++u) {
        int ch = lane + 32 * u;
        h[(size_t)t * CDIM + ch] = (_Float16)((vals[u] - mean) * rs * g[ch] + be[ch]);
    }
}

// ---------------------------------------------------------------------------
// Generic WMMA GEMM: C(M x N) = A(M x K, f16) * Bt^T (Bt is N x K, f16)
// block = 256 threads = 8 waves; wave -> 16 rows x 64 cols (4 wmma per K-step)
// Fragments loaded as contiguous b128 per the gfx1250 f16 VGPR layouts.
// ---------------------------------------------------------------------------
template <int EPI>
__global__ __launch_bounds__(256)
void wmma_gemm(const _Float16* __restrict__ A, const _Float16* __restrict__ Bt,
               const float* __restrict__ bias, int K, int Nout,
               _Float16* __restrict__ outH, float* __restrict__ outF,
               const float* __restrict__ aux) {
    int wave = threadIdx.x >> 5, lane = threadIdx.x & 31;
    int hi = lane >> 4, ln = lane & 15;
    int row0 = blockIdx.y * 128 + wave * 16;
    int col0 = blockIdx.x * 64;

    v8f acc[4] = {};
    const _Float16* Ar = A + (size_t)(row0 + ln) * K + 8 * hi;
    for (int kk = 0; kk < K; kk += 32) {
        Frag af;
        const _Float16* Ap = Ar + kk;
        __builtin_prefetch(Ap + 32, 0, 1);
        af.q[0] = *(const v4u*)(Ap);
        af.q[1] = *(const v4u*)(Ap + 16);
#pragma unroll
        for (int tj = 0; tj < 4; ++tj) {
            Frag bf;
            const _Float16* Bp = Bt + (size_t)(col0 + tj * 16 + ln) * K + kk + 8 * hi;
            bf.q[0] = *(const v4u*)(Bp);
            bf.q[1] = *(const v4u*)(Bp + 16);
            acc[tj] = __builtin_amdgcn_wmma_f32_16x16x32_f16(
                false, af.h, false, bf.h, (short)0, acc[tj], false, false);
        }
    }

#pragma unroll
    for (int tj = 0; tj < 4; ++tj) {
#pragma unroll
        for (int v = 0; v < 8; ++v) {
            int r = row0 + v + 8 * hi;
            int c = col0 + tj * 16 + ln;
            float val = acc[tj][v] + bias[c];
            if (EPI == 0) {
                outH[(size_t)r * Nout + c] = (_Float16)val;
            } else if (EPI == 1) {
                float ge = 0.5f * val * (1.f + erff(val * 0.70710678118654752f));
                outH[(size_t)r * Nout + c] = (_Float16)ge;
            } else if (EPI == 2) {
                // r is a window-ordered token; scatter back with +4 shift, add x
                int n = r & 63, win = (r >> 6) & (NWIN - 1), b = r >> 16;
                int wh = win >> 5, ww = win & 31;
                int i = ((wh << 3) + (n >> 3) + 4) & 255;
                int j = ((ww << 3) + (n & 7) + 4) & 255;
                size_t pix = ((size_t)b << 16) + (size_t)i * 256 + j;
                float sc = aux[((size_t)b * CDIM + c) * 65536 + (size_t)i * 256 + j];
                outF[pix * CDIM + c] = val + sc;
            } else { // EPI == 3
                int b = r >> 16, rem = r & 65535;
                float res = aux[(size_t)r * CDIM + c];
                outF[((size_t)b * CDIM + c) * 65536 + rem] = val + res;
            }
        }
    }
}

// ---------------------------------------------------------------------------
// Window attention: one wave per (window, head), row-tile (ti) streamed to
// keep VGPR pressure < 256 (no s_set_vgpr_msb, good occupancy).
// block = 128 (4 waves = 4 heads), grid = (4096 windows, 2 head-groups)
// qkv layout: [token][576] f16, column = which*192 + head*24 + hd
// ---------------------------------------------------------------------------
__global__ __launch_bounds__(128)
void attn_kernel(const _Float16* __restrict__ qkv, const float* __restrict__ biasTab,
                 _Float16* __restrict__ outO) {
    __shared__ _Float16 ldsP[4 * 16 * 64];    // 8 KB: per-wave 16x64 f16 P tile
#if HAVE_ASYNC_LDS
    __shared__ _Float16 ldsV[64 * 96];        // 12 KB: V slab for this head group
#endif
    int wave = threadIdx.x >> 5, lane = threadIdx.x & 31;
    int hi = lane >> 4, ln = lane & 15;
    int win = blockIdx.x;
    int w = win & (NWIN - 1);
    int wh = w >> 5, ww = w & 31;
    int hh = blockIdx.y * 4 + wave;
    size_t base = (size_t)win * 64 * 576;
    const v4u vzero = {0u, 0u, 0u, 0u};

#if HAVE_ASYNC_LDS
    // Stage V columns [hb, hb+96) of this window into LDS via async DMA.
    {
        int hb = 2 * CDIM + blockIdx.y * 96;   // first V column of this head group
        // 64 rows x 96 halfs = 768 x 16B chunks; 128 threads x 6 chunks
#pragma unroll
        for (int u = 0; u < 6; ++u) {
            int q = threadIdx.x * 6 + u;       // chunk id 0..767
            int tt = q / 12, c = (q - tt * 12) * 8;
            const _Float16* gp = qkv + base + (size_t)tt * 576 + hb + c;
            __builtin_amdgcn_global_load_async_to_lds_b128(
                (as1_v4i)(v4i_gcc*)(void*)gp,
                (as3_v4i)(v4i_gcc*)(void*)&ldsV[tt * 96 + c],
                0, 0);
        }
        __builtin_amdgcn_s_wait_asynccnt(0);
        __syncthreads();
    }
#endif

    // --- K fragments (persistent; hd padded 24 -> 32 with zeros) ---
    Frag kb[4];
#pragma unroll
    for (int tj = 0; tj < 4; ++tj) {
        const _Float16* Kp = qkv + base + (size_t)(tj * 16 + ln) * 576 + CDIM + hh * HD + 8 * hi;
        kb[tj].q[0] = *(const v4u*)(Kp);        // k=8hi..8hi+7 (always < 24)
        v4u t2 = *(const v4u*)(Kp + 16);        // k=16+8hi..23+8hi (pad when hi==1)
        kb[tj].q[1] = hi ? vzero : t2;
    }

    // --- V fragments (persistent): vb[ks][tjh], B-layout over 64 keys ---
    Frag vb[2][2];
#pragma unroll
    for (int ks = 0; ks < 2; ++ks)
#pragma unroll
        for (int tjh = 0; tjh < 2; ++tjh)
#pragma unroll
            for (int v = 0; v < 8; ++v) {
                int kk = 2 * v + 16 * hi;
                int n = tjh * 16 + ln;
                _Float16 e0 = (_Float16)0.f, e1 = (_Float16)0.f;
                if (n < HD) {
#if HAVE_ASYNC_LDS
                    const _Float16* Vp = ldsV + (ks * 32 + kk) * 96 + wave * HD + n;
                    e0 = Vp[0];
                    e1 = Vp[96];
#else
                    const _Float16* Vp = qkv + base + (size_t)(ks * 32 + kk) * 576 +
                                         2 * CDIM + hh * HD + n;
                    e0 = Vp[0];
                    e1 = Vp[576];
#endif
                }
                vb[ks][tjh].e[2 * v]     = e0;
                vb[ks][tjh].e[2 * v + 1] = e1;
            }

    const float scale = 0.20412414523193154f;   // 24^-0.5
    int cls = ((wh == 31) ? 2 : 0) | ((ww == 31) ? 1 : 0);
    const float* bt = biasTab + (((size_t)cls * HEADS + hh) << 12);
    _Float16* P = ldsP + wave * 1024;           // this wave's 16x64 P tile

    // --- stream row tiles: QK^T -> softmax -> P -> P*V -> store O ---
    for (int ti = 0; ti < 4; ++ti) {
        // Q fragment for this row tile
        Frag qa;
        {
            const _Float16* Qp = qkv + base + (size_t)(ti * 16 + ln) * 576 + hh * HD + 8 * hi;
            qa.q[0] = *(const v4u*)(Qp);
            v4u t2 = *(const v4u*)(Qp + 16);    // k=16+8hi..23+8hi (pad when hi==1)
            qa.q[1] = hi ? vzero : t2;
        }

        // S row-block: 4 WMMAs
        v8f s[4];
#pragma unroll
        for (int tj = 0; tj < 4; ++tj) {
            v8f z = {};
            s[tj] = __builtin_amdgcn_wmma_f32_16x16x32_f16(
                false, qa.h, false, kb[tj].h, (short)0, z, false, false);
        }

        // scale + bias table
#pragma unroll
        for (int tj = 0; tj < 4; ++tj)
#pragma unroll
            for (int v = 0; v < 8; ++v) {
                int nq = ti * 16 + v + 8 * hi;
                int nk = tj * 16 + ln;
                s[tj][v] = s[tj][v] * scale + bt[nq * 64 + nk];
            }

        // softmax per row (reduce across tj registers + 16-lane half)
#pragma unroll
        for (int v = 0; v < 8; ++v) {
            float mx = fmaxf(fmaxf(s[0][v], s[1][v]), fmaxf(s[2][v], s[3][v]));
#pragma unroll
            for (int d = 8; d >= 1; d >>= 1) mx = fmaxf(mx, __shfl_xor(mx, d, 32));
            float e0 = __expf(s[0][v] - mx);
            float e1 = __expf(s[1][v] - mx);
            float e2 = __expf(s[2][v] - mx);
            float e3 = __expf(s[3][v] - mx);
            float sum = e0 + e1 + e2 + e3;
#pragma unroll
            for (int d = 8; d >= 1; d >>= 1) sum += __shfl_xor(sum, d, 32);
            float inv = 1.f / sum;
            s[0][v] = e0 * inv; s[1][v] = e1 * inv;
            s[2][v] = e2 * inv; s[3][v] = e3 * inv;
        }

        // stage this 16x64 P tile (per-wave buffer; same-wave DS ops are in-order)
#pragma unroll
        for (int tj = 0; tj < 4; ++tj)
#pragma unroll
            for (int v = 0; v < 8; ++v)
                P[(v + 8 * hi) * 64 + tj * 16 + ln] = (_Float16)s[tj][v];

        // O tile = P * V : 4 WMMAs over two 32-key steps
        v8f o[2] = {};
#pragma unroll
        for (int ks = 0; ks < 2; ++ks) {
            Frag pa;
            const _Float16* Pp = P + ln * 64 + ks * 32 + 16 * hi;
            pa.q[0] = *(const v4u*)(Pp);
            pa.q[1] = *(const v4u*)(Pp + 8);
#pragma unroll
            for (int tjh = 0; tjh < 2; ++tjh)
                o[tjh] = __builtin_amdgcn_wmma_f32_16x16x32_f16(
                    false, pa.h, false, vb[ks][tjh].h, (short)0, o[tjh], false, false);
        }

        // store O rows (hd columns 0..23) as f16 [token][192]
#pragma unroll
        for (int tjh = 0; tjh < 2; ++tjh)
#pragma unroll
            for (int v = 0; v < 8; ++v) {
                int col = tjh * 16 + ln;
                if (col < HD) {
                    int t = win * 64 + ti * 16 + v + 8 * hi;
                    outO[(size_t)t * CDIM + hh * HD + col] = (_Float16)o[tjh][v];
                }
            }
    }
}

// ---------------------------------------------------------------------------
extern "C" void kernel_launch(void* const* d_in, const int* in_sizes, int n_in,
                              void* d_out, int out_size, void* d_ws, size_t ws_size,
                              hipStream_t stream) {
    const float* x       = (const float*)d_in[0];
    const float* norm1_g = (const float*)d_in[1];
    const float* norm1_b = (const float*)d_in[2];
    const float* qkv_w   = (const float*)d_in[3];
    const float* qkv_b   = (const float*)d_in[4];
    const float* proj_w  = (const float*)d_in[5];
    const float* proj_b  = (const float*)d_in[6];
    const float* rpb     = (const float*)d_in[7];
    const float* norm2_g = (const float*)d_in[8];
    const float* norm2_b = (const float*)d_in[9];
    const float* fc1_w   = (const float*)d_in[10];
    const float* fc1_b   = (const float*)d_in[11];
    const float* fc2_w   = (const float*)d_in[12];
    const float* fc2_b   = (const float*)d_in[13];
    float* out = (float*)d_out;

    // ---- workspace partition (byte offsets, 256B aligned) ----
    char* ws = (char*)d_ws;
    size_t off = 0;
    auto alloc = [&](size_t bytes) { void* p = ws + off; off += (bytes + 255) & ~(size_t)255; return p; };
    _Float16* qkvWt  = (_Float16*)alloc((size_t)576 * CDIM * 2);
    _Float16* projWt = (_Float16*)alloc((size_t)CDIM * CDIM * 2);
    _Float16* fc1Wt  = (_Float16*)alloc((size_t)768 * CDIM * 2);
    _Float16* fc2Wt  = (_Float16*)alloc((size_t)CDIM * 768 * 2);
    float*    biasTab = (float*)alloc((size_t)4 * HEADS * 64 * 64 * 4);
    _Float16* bufA   = (_Float16*)alloc((size_t)TOK * CDIM * 2);   // h1, reused as h2
    _Float16* bufQKV = (_Float16*)alloc((size_t)TOK * 576 * 2);
    _Float16* bufO   = (_Float16*)alloc((size_t)TOK * CDIM * 2);   // contiguous w/ bufQKV
    float*    bufXH  = (float*)alloc((size_t)TOK * CDIM * 4);
    _Float16* bufF1  = bufQKV;   // alias: 576+192 = 768 columns; dead before fc1

    // ---- weight converts + bias table ----
    wconv_kernel<<<(CDIM * 576 + 255) / 256, 256, 0, stream>>>(qkv_w, qkvWt, CDIM, 576);
    wconv_kernel<<<(CDIM * CDIM + 255) / 256, 256, 0, stream>>>(proj_w, projWt, CDIM, CDIM);
    wconv_kernel<<<(CDIM * 768 + 255) / 256, 256, 0, stream>>>(fc1_w, fc1Wt, CDIM, 768);
    wconv_kernel<<<(768 * CDIM + 255) / 256, 256, 0, stream>>>(fc2_w, fc2Wt, 768, CDIM);
    bias_prep_kernel<<<(4 * HEADS * 64 * 64) / 256, 256, 0, stream>>>(rpb, biasTab);

    // ---- LN1 + shift + window partition ----
    ln1_kernel<<<TOK / 8, 256, 0, stream>>>(x, norm1_g, norm1_b, bufA);

    // ---- QKV GEMM: (TOK x 192) x (192 x 576) ----
    wmma_gemm<0><<<dim3(576 / 64, TOK / 128), 256, 0, stream>>>(
        bufA, qkvWt, qkv_b, CDIM, 576, bufQKV, nullptr, nullptr);

    // ---- windowed multi-head attention ----
    attn_kernel<<<dim3(4 * NWIN, 2), 128, 0, stream>>>(bufQKV, biasTab, bufO);

    // ---- proj GEMM + window reverse + residual -> xh (B,H,W,C f32) ----
    wmma_gemm<2><<<dim3(CDIM / 64, TOK / 128), 256, 0, stream>>>(
        bufO, projWt, proj_b, CDIM, CDIM, nullptr, bufXH, x);

    // ---- LN2 ----
    ln2_kernel<<<TOK / 8, 256, 0, stream>>>(bufXH, norm2_g, norm2_b, bufA);

    // ---- fc1 GEMM + GELU ----
    wmma_gemm<1><<<dim3(768 / 64, TOK / 128), 256, 0, stream>>>(
        bufA, fc1Wt, fc1_b, CDIM, 768, bufF1, nullptr, nullptr);

    // ---- fc2 GEMM + residual + NHWC->NCHW final store ----
    wmma_gemm<3><<<dim3(CDIM / 64, TOK / 128), 256, 0, stream>>>(
        bufF1, fc2Wt, fc2_b, 768, CDIM, nullptr, out, bufXH);

    (void)in_sizes; (void)n_in; (void)out_size; (void)ws_size;
}
